// LSTM_68272800137597
// MI455X (gfx1250) — matrix operand order repaired and checked by reading, not compiled, not run on hardware
//
#include <hip/hip_runtime.h>

// ---------------------------------------------------------------------------
// Fused 2-layer LSTM (H=16) for MI455X / gfx1250, wave32 WMMA.
//   lstm_in = sigmoid(x @ w_in^T + b_in); h1 = LSTM0(lstm_in); h2 = LSTM1(h1)
//   out = h2 @ w_out^T + b_out
//
// One wave per 16-row batch tile (64 waves), serial over T=1024.
// Per timestep: 1 WMMA input proj (K=32=D_in) + 4 WMMAs layer0 gates
// (A = [lstm_in | h0], K=16+16=32) + 4 WMMAs layer1 + 1 WMMA output proj.
// Biases folded into the C accumulator. h state transposed D->A layout via
// LDS (16x16 f16 tiles, row stride 48B => conflict-free ds_load_b128).
// Activations use v_exp_f32 + v_rcp_f32 (single TRANS ops) instead of the
// IEEE division expansion, shortening the latency-critical per-step chain.
// HBM traffic = read x once (128MB) + write out once (32MB); everything else
// stays in registers/LDS.
// ---------------------------------------------------------------------------

typedef __attribute__((ext_vector_type(16))) _Float16 v16h;
typedef __attribute__((ext_vector_type(8)))  _Float16 v8h;
typedef __attribute__((ext_vector_type(8)))  float    v8f;

#define TSTEPS 1024
#define BATCH  1024
#define DIN    32
#define HID    16
#define DOUT   8
#define LROWS  24   // LDS row stride in halves (48 bytes: 16B aligned, no bank conflicts)

// Single-instruction hardware reciprocal (v_rcp_f32): avoids the
// div_scale/rcp/fma Newton chain the precise "/" operator generates.
__device__ __forceinline__ float fast_rcp(float x) { return __builtin_amdgcn_rcpf(x); }
// sigmoid(x) = 1/(1+e^-x)  -> v_exp + v_add + v_rcp
__device__ __forceinline__ float fsigmoid(float x) { return fast_rcp(1.0f + __expf(-x)); }
// tanh(x) = 1 - 2/(e^{2x}+1) -> v_exp + v_add + v_rcp + fma
__device__ __forceinline__ float ftanh_fast(float x) {
  return 1.0f - 2.0f * fast_rcp(__expf(2.0f * x) + 1.0f);
}

__device__ __forceinline__ v8f bcast8(float v) {
  v8f r;
#pragma unroll
  for (int i = 0; i < 8; ++i) r[i] = v;
  return r;
}

__global__ __launch_bounds__(32)
void lstm_fused_kernel(const float* __restrict__ x,
                       const float* __restrict__ w_in,  const float* __restrict__ b_in,
                       const float* __restrict__ w_ih0, const float* __restrict__ w_hh0,
                       const float* __restrict__ b_ih0, const float* __restrict__ b_hh0,
                       const float* __restrict__ w_ih1, const float* __restrict__ w_hh1,
                       const float* __restrict__ b_ih1, const float* __restrict__ b_hh1,
                       const float* __restrict__ w_out, const float* __restrict__ b_out,
                       float* __restrict__ out)
{
  __shared__ alignas(16) _Float16 lds_in[16 * LROWS];
  __shared__ alignas(16) _Float16 lds_h0[16 * LROWS];
  __shared__ alignas(16) _Float16 lds_h1[16 * LROWS];

  const int lane  = threadIdx.x;     // 0..31
  const int m     = lane & 15;       // row (A) / column (B,C,D) index within tile
  const int half  = lane >> 4;       // 0/1: which K-half (A) / K-range (B)
  const int brow0 = blockIdx.x * 16; // first batch row of this wave's tile

  // zero initial h0 / h1 state (16*24 = 384 halves, 12 per lane, uniform trip count)
  for (int i = lane; i < 16 * LROWS; i += 32) {
    lds_h0[i] = (_Float16)0.f;
    lds_h1[i] = (_Float16)0.f;
  }

  // ---- B fragments (K x N, f16). lane&15 = N column; half h: K=16h..16h+15 (Bin), else K=8h..8h+7,16+8h..16+8h+7
  v16h Bin;  // w_in^T : [K=32 x N=16], Bin[j] = w_in[n][16*half + j]
  {
    const float* wr = w_in + m * DIN + half * 16;
#pragma unroll
    for (int j = 0; j < 16; ++j) Bin[j] = (_Float16)wr[j];
  }
  // Gate weights, K = [x-part(16) | h-part(16)]: elems 0-7 = w_ih cols 8h..8h+7, elems 8-15 = w_hh cols 8h..8h+7
  v16h B0[4], B1[4];
#pragma unroll
  for (int g = 0; g < 4; ++g) {               // torch gate order: i, f, g, o
    const int n = g * 16 + m;
    const float* s0 = w_ih0 + n * HID + 8 * half;
    const float* r0 = w_hh0 + n * HID + 8 * half;
    const float* s1 = w_ih1 + n * HID + 8 * half;
    const float* r1 = w_hh1 + n * HID + 8 * half;
#pragma unroll
    for (int j = 0; j < 16; ++j) {
      B0[g][j] = (_Float16)(j < 8 ? s0[j] : r0[j - 8]);
      B1[g][j] = (_Float16)(j < 8 ? s1[j] : r1[j - 8]);
    }
  }
  v16h Bo;   // w_out^T padded: valid only for N<8, K<16
  {
    const bool live = (m < DOUT);
    const float* wr = w_out + m * HID + 8 * half;
#pragma unroll
    for (int j = 0; j < 16; ++j) Bo[j] = (live && j < 8) ? (_Float16)wr[j] : (_Float16)0.f;
  }

  // ---- biases folded into C accumulators (C/D layout: N = lane&15, broadcast over 8 M rows)
  const v8f Cin = bcast8(b_in[m]);
  v8f Cg0[4], Cg1[4];
#pragma unroll
  for (int g = 0; g < 4; ++g) {
    Cg0[g] = bcast8(b_ih0[g * 16 + m] + b_hh0[g * 16 + m]);
    Cg1[g] = bcast8(b_ih1[g * 16 + m] + b_hh1[g * 16 + m]);
  }
  const v8f Cout = bcast8((m < DOUT) ? b_out[m] : 0.f);

  v8f c0 = bcast8(0.f), c1 = bcast8(0.f);  // cell states, D layout
  const v8h zero8 = {};

  __syncthreads();

  const int ldsA = m * LROWS + 8 * half;   // per-lane A-fragment LDS row offset (16B aligned)

  for (int t = 0; t < TSTEPS; ++t) {
    // ================= input projection: lstm_in = sigmoid(x_t @ w_in^T + b_in)
    // A fragment of x_t (16x32 f32 -> f16). Lane holds row m, K = {16h..16h+15}
    const float* xr = x + ((size_t)t * BATCH + brow0 + m) * DIN + half * 16;
    if (t + 1 < TSTEPS) __builtin_prefetch(xr + (size_t)BATCH * DIN, 0, 0);
    v16h Ax;
#pragma unroll
    for (int j = 0; j < 8; ++j) {
      Ax[j]     = (_Float16)xr[j];
      Ax[j + 8] = (_Float16)xr[8 + j];
    }
    v8f din = __builtin_amdgcn_wmma_f32_16x16x32_f16(false, Ax, false, Bin, (short)0, Cin, false, false);
#pragma unroll
    for (int r = 0; r < 8; ++r)
      lds_in[(r + 8 * half) * LROWS + m] = (_Float16)fsigmoid(din[r]);
    __syncthreads();

    // ================= layer 0: gates = [lstm_in | h0_prev] @ [w_ih0 ; w_hh0]^T + b
    {
      v8h a_lo = *(const v8h*)&lds_in[ldsA];   // K-part 0..15  (input)
      v8h a_hi = *(const v8h*)&lds_h0[ldsA];   // K-part 16..31 (recurrent h0)
      v16h A0 = __builtin_shufflevector(a_lo, a_hi, 0,1,2,3,4,5,6,7,8,9,10,11,12,13,14,15);
      v8f gi = __builtin_amdgcn_wmma_f32_16x16x32_f16(false, A0, false, B0[0], (short)0, Cg0[0], false, false);
      v8f gf = __builtin_amdgcn_wmma_f32_16x16x32_f16(false, A0, false, B0[1], (short)0, Cg0[1], false, false);
      v8f gg = __builtin_amdgcn_wmma_f32_16x16x32_f16(false, A0, false, B0[2], (short)0, Cg0[2], false, false);
      v8f go = __builtin_amdgcn_wmma_f32_16x16x32_f16(false, A0, false, B0[3], (short)0, Cg0[3], false, false);
#pragma unroll
      for (int r = 0; r < 8; ++r) {
        float iv = fsigmoid(gi[r]);
        float fv = fsigmoid(gf[r]);
        float gv = ftanh_fast(gg[r]);
        float ov = fsigmoid(go[r]);
        float cn = fv * c0[r] + iv * gv;
        c0[r] = cn;
        lds_h0[(r + 8 * half) * LROWS + m] = (_Float16)(ov * ftanh_fast(cn));
      }
    }
    __syncthreads();

    // ================= layer 1: gates = [h0_t | h1_prev] @ [w_ih1 ; w_hh1]^T + b
    {
      v8h a_lo = *(const v8h*)&lds_h0[ldsA];   // layer-0 output at time t
      v8h a_hi = *(const v8h*)&lds_h1[ldsA];   // recurrent h1 (t-1)
      v16h A1 = __builtin_shufflevector(a_lo, a_hi, 0,1,2,3,4,5,6,7,8,9,10,11,12,13,14,15);
      v8f gi = __builtin_amdgcn_wmma_f32_16x16x32_f16(false, A1, false, B1[0], (short)0, Cg1[0], false, false);
      v8f gf = __builtin_amdgcn_wmma_f32_16x16x32_f16(false, A1, false, B1[1], (short)0, Cg1[1], false, false);
      v8f gg = __builtin_amdgcn_wmma_f32_16x16x32_f16(false, A1, false, B1[2], (short)0, Cg1[2], false, false);
      v8f go = __builtin_amdgcn_wmma_f32_16x16x32_f16(false, A1, false, B1[3], (short)0, Cg1[3], false, false);
#pragma unroll
      for (int r = 0; r < 8; ++r) {
        float iv = fsigmoid(gi[r]);
        float fv = fsigmoid(gf[r]);
        float gv = ftanh_fast(gg[r]);
        float ov = fsigmoid(go[r]);
        float cn = fv * c1[r] + iv * gv;
        c1[r] = cn;
        lds_h1[(r + 8 * half) * LROWS + m] = (_Float16)(ov * ftanh_fast(cn));
      }
    }
    __syncthreads();

    // ================= output projection: out_t = h1_t @ w_out^T + b_out  (K=16, padded)
    {
      v8h h_lo = *(const v8h*)&lds_h1[ldsA];
      v16h Ao = __builtin_shufflevector(h_lo, zero8, 0,1,2,3,4,5,6,7,8,9,10,11,12,13,14,15);
      v8f dout = __builtin_amdgcn_wmma_f32_16x16x32_f16(false, Ao, false, Bo, (short)0, Cout, false, false);
      if (m < DOUT) {
        float* orow = out + ((size_t)t * BATCH + brow0 + 8 * half) * DOUT + m;
#pragma unroll
        for (int r = 0; r < 8; ++r) orow[r * DOUT] = dout[r];
      }
    }
  }
}

extern "C" void kernel_launch(void* const* d_in, const int* in_sizes, int n_in,
                              void* d_out, int out_size, void* d_ws, size_t ws_size,
                              hipStream_t stream) {
  const float* x     = (const float*)d_in[0];
  const float* w_in  = (const float*)d_in[1];
  const float* b_in  = (const float*)d_in[2];
  const float* w_ih0 = (const float*)d_in[3];
  const float* w_hh0 = (const float*)d_in[4];
  const float* b_ih0 = (const float*)d_in[5];
  const float* b_hh0 = (const float*)d_in[6];
  const float* w_ih1 = (const float*)d_in[7];
  const float* w_hh1 = (const float*)d_in[8];
  const float* b_ih1 = (const float*)d_in[9];
  const float* b_hh1 = (const float*)d_in[10];
  const float* w_out = (const float*)d_in[11];
  const float* b_out = (const float*)d_in[12];
  float* out = (float*)d_out;

  // one wave (32 threads) per 16-row batch tile; 1024/16 = 64 workgroups
  lstm_fused_kernel<<<dim3(BATCH / 16), dim3(32), 0, stream>>>(
      x, w_in, b_in, w_ih0, w_hh0, b_ih0, b_hh0,
      w_ih1, w_hh1, b_ih1, b_hh1, w_out, b_out, out);
}
